// ChordAwareAttention_74912819577350
// MI455X (gfx1250) — compile-verified, matrix operand order checked by reading, not run on hardware
//
#include <hip/hip_runtime.h>
#include <hip/hip_bf16.h>

// ---------------------------------------------------------------------------
// ChordAwareAttention on MI455X (gfx1250): all big matmuls via
// v_wmma_f32_16x16x32_f16 (f16 operands, f32 accumulation).
// B=2, S=2048, H=1024, NH=16, HD=64; q_cat/k_cat feature dim = 5*HD = 320.
// ---------------------------------------------------------------------------

#define BB 2
#define SS 2048
#define HH 1024
#define NHH 16
#define HDD 64
#define DCAT 320            // 5 * HD

typedef __attribute__((ext_vector_type(16))) _Float16 v16h;
typedef __attribute__((ext_vector_type(8)))  _Float16 v8h;
typedef __attribute__((ext_vector_type(8)))  float    v8f;
typedef __attribute__((ext_vector_type(4)))  float    v4f;

// ---- WMMA fragment loaders --------------------------------------------------
// 16-bit A-matrix 16x32 layout (ISA 7.12.2): lane = m (0..15) + 16*hi.
//   hi=0: elems 0..7 -> K=kk+0..7,  elems 8..15 -> K=kk+16..23
//   hi=1: elems 0..7 -> K=kk+8..15, elems 8..15 -> K=kk+24..31
// B mirrors this per-column (lane n holds column n of the 32x16 B tile), so a
// row-major [N,K] operand loads with the identical pattern.

__device__ __forceinline__ v16h frag_from_h(const _Float16* __restrict__ p, int hi) {
  v8h a = *(const v8h*)(p + 8 * hi);
  v8h b = *(const v8h*)(p + 16 + 8 * hi);
  v16h f;
#pragma unroll
  for (int i = 0; i < 8; ++i) { f[i] = a[i]; f[i + 8] = b[i]; }
  return f;
}

__device__ __forceinline__ v16h frag_from_f(const float* __restrict__ p, int hi) {
  v4f a0 = *(const v4f*)(p + 8 * hi);
  v4f a1 = *(const v4f*)(p + 8 * hi + 4);
  v4f b0 = *(const v4f*)(p + 16 + 8 * hi);
  v4f b1 = *(const v4f*)(p + 16 + 8 * hi + 4);
  v16h f;
#pragma unroll
  for (int i = 0; i < 4; ++i) {
    f[i]      = (_Float16)a0[i];
    f[i + 4]  = (_Float16)a1[i];
    f[i + 8]  = (_Float16)b0[i];
    f[i + 12] = (_Float16)b1[i];
  }
  return f;
}

#define WMMA_F16(a, b, c) \
  __builtin_amdgcn_wmma_f32_16x16x32_f16(false, (a), false, (b), (short)0, (c), false, false)

// ---- weight transpose + f32->f16 convert: W[K,N] -> Wt[N,K] ----------------
// 32x32 LDS tile transpose: both global streams fully coalesced.
__global__ __launch_bounds__(256) void transpose_w_kernel(
    const float* __restrict__ W, _Float16* __restrict__ Wt) {
  __shared__ _Float16 tile[32][33];
  int bk = blockIdx.x & 31;   // k tile
  int bn = blockIdx.x >> 5;   // n tile
  int tx = threadIdx.x & 31;
  int ty = threadIdx.x >> 5;  // 0..7
#pragma unroll
  for (int i = 0; i < 32; i += 8) {
    int k = bk * 32 + ty + i;
    int n = bn * 32 + tx;
    tile[ty + i][tx] = (_Float16)W[(size_t)k * 1024 + n];
  }
  __syncthreads();
#pragma unroll
  for (int i = 0; i < 32; i += 8) {
    int n = bn * 32 + ty + i;
    int k = bk * 32 + tx;
    Wt[(size_t)n * 1024 + k] = tile[tx][ty + i];
  }
}

// ---- big projection GEMM: C[4096,1024] = A_f32 @ Wt^T + bias ---------------
// Epilogue scatters into head-split layouts:
//   vmode==0: dst[((b*NH+h)*S + s)*dstld + aofs + j]   (Qcat / Kcat, f16)
//   vmode==1: dst[((b*NH+h)*HD + j)*S + s]             (V transposed, f16)
__global__ __launch_bounds__(256) void proj_gemm_kernel(
    const float* __restrict__ A, const _Float16* __restrict__ Bt,
    const float* __restrict__ bias, _Float16* __restrict__ dst,
    int dstld, int aofs, int vmode) {
  int lane = threadIdx.x & 31, wave = threadIdx.x >> 5;
  int job = blockIdx.x * 8 + wave;          // 4096 jobs == grid exactly
  int mt = job >> 4;                        // 0..255 row tiles of 16
  int ng = job & 15;                        // group of 64 output cols
  int lr = lane & 15, hi = lane >> 4;

  const float* arow = A + (size_t)(mt * 16 + lr) * HH;
  v8f acc[4] = {};
  for (int kk = 0; kk < HH; kk += 32) {
    v16h a = frag_from_f(arow + kk, hi);
#pragma unroll
    for (int t = 0; t < 4; ++t) {
      const _Float16* brow = Bt + (size_t)(ng * 64 + t * 16 + lr) * HH + kk;
      v16h b = frag_from_h(brow, hi);
      acc[t] = WMMA_F16(a, b, acc[t]);
    }
  }
#pragma unroll
  for (int t = 0; t < 4; ++t) {
    int c = ng * 64 + t * 16 + lr;          // global output column
    int h = c >> 6, j = c & 63;
    float bv = bias[c];
#pragma unroll
    for (int r = 0; r < 8; ++r) {
      int row = mt * 16 + r + 8 * hi;
      int b_ = row >> 11, s = row & (SS - 1);
      float v = acc[t][r] + bv;
      size_t idx;
      if (vmode)
        idx = ((size_t)(b_ * NHH + h) * HDD + j) * SS + s;
      else
        idx = ((size_t)(b_ * NHH + h) * SS + s) * (size_t)dstld + aofs + j;
      dst[idx] = (_Float16)v;
    }
  }
}

// ---- small feature projections (K<=24): VALU, broadcast over batch ---------
__global__ __launch_bounds__(256) void feat_proj_kernel(
    const float* __restrict__ F, int Kf, const float* __restrict__ W,
    const float* __restrict__ bias, _Float16* __restrict__ Kcat, int aofs) {
  int tid = blockIdx.x * 256 + threadIdx.x;   // 0 .. S*H-1
  int c = tid & 1023, s = tid >> 10;
  float acc = bias[c];
  for (int k = 0; k < Kf; ++k) acc += F[s * Kf + k] * W[k * 1024 + c];
  int h = c >> 6, j = c & 63;
  _Float16 v = (_Float16)acc;
#pragma unroll
  for (int b = 0; b < BB; ++b)
    Kcat[((size_t)(b * NHH + h) * SS + s) * DCAT + aofs + j] = v;
}

// ---- scores + exact softmax ------------------------------------------------
// One workgroup (8 wave32) owns a 16-row x 2048-col score stripe in LDS
// (16*2048*4 = 128 KB; CDNA5 WGP has 320 KB). Q fragments for the stripe
// (10 x v16h = 80 VGPRs) are hoisted into registers once per wave; each wave
// then sweeps 16 column tiles streaming only K fragments. The block finishes
// with an exact softmax in LDS and streams normalized probs out as float4.
__global__ __launch_bounds__(256) void scores_softmax_kernel(
    const _Float16* __restrict__ Qcat, const _Float16* __restrict__ Kcat,
    float* __restrict__ attn) {
  extern __shared__ float smem[];
  float* sc = smem;                 // 16 * 2048
  float* red = smem + 16 * SS;      // 256
  float* rowstat = red + 256;       // [0..15] = max, [16..31] = 1/sum

  int tid = threadIdx.x;
  int lane = tid & 31, wave = tid >> 5;
  int bh = blockIdx.x >> 7;         // 0..31
  int mt = blockIdx.x & 127;        // row tile
  int lr = lane & 15, hi = lane >> 4;

  const _Float16* qrow = Qcat + ((size_t)bh * SS + mt * 16 + lr) * DCAT;
  v16h qf[10];
#pragma unroll
  for (int kt = 0; kt < 10; ++kt) qf[kt] = frag_from_h(qrow + kt * 32, hi);

  for (int ct = 0; ct < 16; ++ct) {
    int col0 = wave * 256 + ct * 16;
    const _Float16* krow = Kcat + ((size_t)bh * SS + col0 + lr) * DCAT;
    v8f acc = {};
#pragma unroll
    for (int kt = 0; kt < 10; ++kt) {
      v16h b = frag_from_h(krow + kt * 32, hi);
      acc = WMMA_F16(qf[kt], b, acc);
    }
#pragma unroll
    for (int r = 0; r < 8; ++r)
      sc[(r + 8 * hi) * SS + col0 + lr] = acc[r] * 0.025f;  // (HD^-0.5)/5
  }
  __syncthreads();

  int row = tid >> 4, seg = tid & 15;
  float m = -1e30f;
  for (int c = seg; c < SS; c += 16) m = fmaxf(m, sc[row * SS + c]);
  red[tid] = m;
  __syncthreads();
  if (seg == 0) {
    float mm = red[row * 16];
    for (int i = 1; i < 16; ++i) mm = fmaxf(mm, red[row * 16 + i]);
    rowstat[row] = mm;
  }
  __syncthreads();
  float rm = rowstat[row];
  float sum = 0.f;
  for (int c = seg; c < SS; c += 16) {
    float e = __expf(sc[row * SS + c] - rm);
    sc[row * SS + c] = e;                    // store exp back (reused below)
    sum += e;
  }
  red[tid] = sum;
  __syncthreads();
  if (seg == 0) {
    float ss = 0.f;
    for (int i = 0; i < 16; ++i) ss += red[row * 16 + i];
    rowstat[16 + row] = 1.0f / ss;
  }
  __syncthreads();

  // normalized write, float4-coalesced (attn base is 16 MiB-offset -> aligned)
  float4* attn4 = (float4*)(attn + ((size_t)bh * SS + mt * 16) * SS);
  const float4* sc4 = (const float4*)sc;
  for (int idx = tid; idx < 16 * SS / 4; idx += 256) {
    int r2 = idx >> 9;                       // 512 float4 per row
    float inv = rowstat[16 + r2];
    float4 e = sc4[idx];
    e.x *= inv; e.y *= inv; e.z *= inv; e.w *= inv;
    attn4[idx] = e;
  }
}

// ---- ctx = attn @ V  (per (b,h): [S,S] x [S,64]) ---------------------------
__global__ __launch_bounds__(256) void av_gemm_kernel(
    const float* __restrict__ attn, const _Float16* __restrict__ Vt,
    _Float16* __restrict__ ctx) {
  int lane = threadIdx.x & 31, wave = threadIdx.x >> 5;
  int job = blockIdx.x * 8 + wave;          // 4096 jobs
  int bh = job >> 7;                        // 0..31
  int mt = job & 127;
  int lr = lane & 15, hi = lane >> 4;

  const float* arow = attn + ((size_t)bh * SS + mt * 16 + lr) * SS;
  v8f acc[4] = {};
  for (int kk = 0; kk < SS; kk += 32) {
    v16h a = frag_from_f(arow + kk, hi);
#pragma unroll
    for (int t = 0; t < 4; ++t) {
      const _Float16* brow = Vt + ((size_t)bh * HDD + t * 16 + lr) * SS + kk;
      v16h b = frag_from_h(brow, hi);
      acc[t] = WMMA_F16(a, b, acc[t]);
    }
  }
  int b_ = bh >> 4, h = bh & 15;
#pragma unroll
  for (int t = 0; t < 4; ++t) {
    int c = h * HDD + t * 16 + lr;          // column inside H
#pragma unroll
    for (int r = 0; r < 8; ++r) {
      int s = mt * 16 + r + 8 * hi;
      ctx[((size_t)(b_ * SS + s)) * HH + c] = (_Float16)acc[t][r];
    }
  }
}

// ---- out = ctx @ Wout + bias (f32 result) ----------------------------------
__global__ __launch_bounds__(256) void out_gemm_kernel(
    const _Float16* __restrict__ A, const _Float16* __restrict__ Bt,
    const float* __restrict__ bias, float* __restrict__ out) {
  int lane = threadIdx.x & 31, wave = threadIdx.x >> 5;
  int job = blockIdx.x * 8 + wave;          // 4096 jobs
  int mt = job >> 4;
  int ng = job & 15;
  int lr = lane & 15, hi = lane >> 4;

  const _Float16* arow = A + (size_t)(mt * 16 + lr) * HH;
  v8f acc[4] = {};
  for (int kk = 0; kk < HH; kk += 32) {
    v16h a = frag_from_h(arow + kk, hi);
#pragma unroll
    for (int t = 0; t < 4; ++t) {
      const _Float16* brow = Bt + (size_t)(ng * 64 + t * 16 + lr) * HH + kk;
      v16h b = frag_from_h(brow, hi);
      acc[t] = WMMA_F16(a, b, acc[t]);
    }
  }
#pragma unroll
  for (int t = 0; t < 4; ++t) {
    int c = ng * 64 + t * 16 + lr;
    float bv = bias[c];
#pragma unroll
    for (int r = 0; r < 8; ++r) {
      int row = mt * 16 + r + 8 * hi;
      out[(size_t)row * HH + c] = acc[t][r] + bv;
    }
  }
}

// ---------------------------------------------------------------------------
extern "C" void kernel_launch(void* const* d_in, const int* in_sizes, int n_in,
                              void* d_out, int out_size, void* d_ws, size_t ws_size,
                              hipStream_t stream) {
  (void)in_sizes; (void)n_in; (void)out_size; (void)ws_size;

  const size_t W1M = (size_t)1024 * 1024;

  // workspace layout (f16), ~112 MiB total
  char* ws = (char*)d_ws;
  _Float16* Wt   = (_Float16*)ws;                                    // 8 x 1M
  _Float16* Qcat = (_Float16*)(ws + 8 * W1M * 2);                    // B*NH*S*320
  _Float16* Kcat = (_Float16*)(ws + 8 * W1M * 2 + 41943040);
  _Float16* Vt   = (_Float16*)(ws + 8 * W1M * 2 + 2 * 41943040ull);  // B*NH*64*S
  _Float16* ctx  = (_Float16*)(ws + 8 * W1M * 2 + 2 * 41943040ull + 8388608);

  float* out  = (float*)d_out;                       // [B,S,H]
  float* attn = out + (size_t)BB * SS * HH;          // [B,NH,S,S]

  // weight order: pitch, harmony, melody, voice, rhythm, key, value, out
  const int wIdx[8] = {9, 11, 15, 13, 17, 19, 21, 35};
  const int bIdx[8] = {10, 12, 16, 14, 18, 20, 22, 36};
  for (int i = 0; i < 8; ++i)
    transpose_w_kernel<<<1024, 256, 0, stream>>>((const float*)d_in[wIdx[i]],
                                                 Wt + (size_t)i * W1M);

  // 5 query-aspect projections into Qcat (aofs = aspect*64)
  for (int i = 0; i < 5; ++i)
    proj_gemm_kernel<<<512, 256, 0, stream>>>(
        (const float*)d_in[0], Wt + (size_t)i * W1M, (const float*)d_in[bIdx[i]],
        Qcat, DCAT, i * 64, 0);
  // key -> Kcat[..,0:64]
  proj_gemm_kernel<<<512, 256, 0, stream>>>(
      (const float*)d_in[1], Wt + 5 * W1M, (const float*)d_in[20], Kcat, DCAT, 0, 0);
  // value -> Vt (transposed per head)
  proj_gemm_kernel<<<512, 256, 0, stream>>>(
      (const float*)d_in[2], Wt + 6 * W1M, (const float*)d_in[22], Vt, 0, 0, 1);

  // feature projections into Kcat: chord@64, melody@128, bass@192, rhythm@256
  feat_proj_kernel<<<8192, 256, 0, stream>>>((const float*)d_in[3], 12,
      (const float*)d_in[23], (const float*)d_in[24], Kcat, 64);
  feat_proj_kernel<<<8192, 256, 0, stream>>>((const float*)d_in[5], 24,
      (const float*)d_in[27], (const float*)d_in[28], Kcat, 128);
  feat_proj_kernel<<<8192, 256, 0, stream>>>((const float*)d_in[4], 12,
      (const float*)d_in[25], (const float*)d_in[26], Kcat, 192);
  feat_proj_kernel<<<8192, 256, 0, stream>>>((const float*)d_in[8], 4,
      (const float*)d_in[33], (const float*)d_in[34], Kcat, 256);

  // scores + softmax -> attn (f32, second output)
  const int smemBytes = (16 * SS + 256 + 32) * (int)sizeof(float);  // 132224 B
  scores_softmax_kernel<<<4096, 256, smemBytes, stream>>>(Qcat, Kcat, attn);

  // ctx = attn @ V
  av_gemm_kernel<<<512, 256, 0, stream>>>(attn, Vt, ctx);

  // out = ctx @ Wout + b
  out_gemm_kernel<<<512, 256, 0, stream>>>(ctx, Wt + 7 * W1M,
                                           (const float*)d_in[36], out);
}
